// HybridHeadAttention_65128884077009
// MI455X (gfx1250) — matrix operand order repaired and checked by reading, not compiled, not run on hardware
//
#include <hip/hip_runtime.h>
#include <hip/hip_bf16.h>

// ---------------- types / constants ----------------
typedef __bf16 bf16;
typedef __attribute__((ext_vector_type(16))) __bf16 bf16x16;
typedef __attribute__((ext_vector_type(8)))  __bf16 bf16x8;
typedef __attribute__((ext_vector_type(8)))  float  f32x8;

#define DIMC   768
#define NHEAD  12
#define HD     64
#define NTOK   4096
#define BATCH  4
#define MLM    64
#define PNP    11
#define BHP    (BATCH*PNP)      // 44
#define BN     (BATCH*NTOK)     // 16384
#define QKVN   (3*DIMC)         // 2304
#define QSCALE 0.125f           // 64^-0.5

__device__ __forceinline__ f32x8 zero8() {
  f32x8 z;
#pragma unroll
  for (int i = 0; i < 8; ++i) z[i] = 0.0f;
  return z;
}

__device__ __forceinline__ f32x8 wmma_bf16(bf16x16 a, bf16x16 b, f32x8 c) {
  // emits v_wmma_f32_16x16x32_bf16
  return __builtin_amdgcn_wmma_f32_16x16x32_bf16(false, a, false, b, (short)0, c,
                                                 false, false);
}

// A-fragment (16x32 bf16, M x K): lane holds row m=lane&15; elements 0..15 at
// K = half*16 .. +15 (contiguous, matches B fragments)
__device__ __forceinline__ bf16x16 load_a_frag(const bf16* p) {
  bf16x8 lo = *(const bf16x8*)p;
  bf16x8 hi = *(const bf16x8*)(p + 8);
  bf16x16 r;
#pragma unroll
  for (int i = 0; i < 8; ++i) { r[i] = lo[i]; r[i + 8] = hi[i]; }
  return r;
}

// One wave computes a 16(M) x 64(N) tile of  C = A[M,K] * B[N,K]^T, K = 32*KSTEPS.
// Software-pipelined: step k+1's fragment loads are issued before step k's WMMAs,
// so the scheduler can overlap global_load_b128 latency with matrix work
// (partial s_wait_loadcnt instead of wait-to-zero).
template <int KSTEPS>
__device__ __forceinline__ void wmma_rowtile(const bf16* __restrict__ A, int lda,
                                             const bf16* __restrict__ B, int ldb,
                                             f32x8 acc[4]) {
  const int lane = threadIdx.x & 31;
  const int hl   = lane >> 4;
  const int m    = lane & 15;
  const bf16* ap = A + (size_t)m * lda + hl * 16;
  const bf16* bp = B + (size_t)m * ldb + hl * 16;
  // stage-0 preload
  bf16x8 alo = *(const bf16x8*)ap;
  bf16x8 ahi = *(const bf16x8*)(ap + 8);
  bf16x16 bq[4];
#pragma unroll
  for (int t = 0; t < 4; ++t) bq[t] = *(const bf16x16*)(bp + (size_t)(16 * t) * ldb);
#pragma unroll
  for (int ks = 0; ks < KSTEPS; ++ks) {
    bf16x16 a;
#pragma unroll
    for (int i = 0; i < 8; ++i) { a[i] = alo[i]; a[i + 8] = ahi[i]; }
    bf16x16 bn[4];
#pragma unroll
    for (int t = 0; t < 4; ++t) bn[t] = bq[t];
    if (ks + 1 < KSTEPS) {                       // prefetch next k-step into regs
      const int k0 = (ks + 1) * 32;
      alo = *(const bf16x8*)(ap + k0);
      ahi = *(const bf16x8*)(ap + k0 + 8);
#pragma unroll
      for (int t = 0; t < 4; ++t)
        bq[t] = *(const bf16x16*)(bp + (size_t)(16 * t) * ldb + k0);
    }
#pragma unroll
    for (int t = 0; t < 4; ++t) acc[t] = wmma_bf16(a, bn[t], acc[t]);
  }
}

// ---------------- kernels ----------------

__global__ void k_cvt_bf16(const float* __restrict__ in, bf16* __restrict__ out, int n) {
  int i = blockIdx.x * blockDim.x + threadIdx.x;
  if (i < n) out[i] = (bf16)in[i];
}

// qkv = x @ qkv_w^T + b ; scatter into per-head q(scaled)/k/v bf16, vT for head 11
__global__ void k_qkv_gemm(const bf16* __restrict__ X, const bf16* __restrict__ W,
                           const float* __restrict__ bias,
                           bf16* __restrict__ qb, bf16* __restrict__ kb,
                           bf16* __restrict__ vb, bf16* __restrict__ vt) {
  const int wave = threadIdx.x >> 5;
  const int lane = threadIdx.x & 31, hl = lane >> 4, m = lane & 15;
  const int rowbase = blockIdx.x * 16;
  const int colbase = blockIdx.y * 256 + wave * 64;
  // pull upcoming weight panel lines toward L2 (global_prefetch_b8)
  __builtin_prefetch(W + (size_t)(colbase + lane) * DIMC, 0, 1);
  __builtin_prefetch(W + (size_t)(colbase + 32 + lane) * DIMC, 0, 1);
  f32x8 acc[4];
#pragma unroll
  for (int t = 0; t < 4; ++t) acc[t] = zero8();
  wmma_rowtile<DIMC / 32>(X + (size_t)rowbase * DIMC, DIMC,
                          W + (size_t)colbase * DIMC, DIMC, acc);
#pragma unroll
  for (int t = 0; t < 4; ++t) {
    int col = colbase + 16 * t + m;
    int which = col / DIMC;
    int rem = col - which * DIMC;
    int h = rem >> 6, d = rem & 63;
    float bs = bias[col];
#pragma unroll
    for (int v = 0; v < 8; ++v) {
      int row = rowbase + v + 8 * hl;
      int b = row >> 12, n = row & 4095;
      float val = acc[t][v] + bs;
      size_t idx = (((size_t)b * NHEAD + h) * NTOK + n) * HD + d;
      if (which == 0)      qb[idx] = (bf16)(val * QSCALE);
      else if (which == 1) kb[idx] = (bf16)val;
      else {
        vb[idx] = (bf16)val;
        if (h == 11) vt[((size_t)b * HD + d) * NTOK + n] = (bf16)val;  // V^T for flash head
      }
    }
  }
}

// SA[bh] = qp(4096x64) @ k_m(64x64)^T   (landmark cols: row stride 64*HD)
__global__ void k_sa_gemm(const bf16* __restrict__ qb, const bf16* __restrict__ kb,
                          float* __restrict__ SA) {
  const int bh = blockIdx.x, b = bh / PNP, h = bh % PNP;
  const int wave = threadIdx.x >> 5;
  const int lane = threadIdx.x & 31, hl = lane >> 4, m = lane & 15;
  const int rowbase = blockIdx.y * 64 + wave * 16;
  const bf16* A = qb + (((size_t)b * NHEAD + h) * NTOK + rowbase) * HD;
  const bf16* B = kb + ((size_t)b * NHEAD + h) * NTOK * HD;   // landmark n -> row n*64
  f32x8 acc[4];
#pragma unroll
  for (int t = 0; t < 4; ++t) acc[t] = zero8();
  wmma_rowtile<2>(A, HD, B, HD * 64, acc);
#pragma unroll
  for (int t = 0; t < 4; ++t)
#pragma unroll
    for (int v = 0; v < 8; ++v)
      SA[((size_t)bh * NTOK + rowbase + v + 8 * hl) * MLM + 16 * t + m] = acc[t][v];
}

// SC[bh] = q_m(64x64) @ kp(4096x64)^T   (landmark rows: stride 64*HD)
__global__ void k_sc_gemm(const bf16* __restrict__ qb, const bf16* __restrict__ kb,
                          float* __restrict__ SC) {
  const int bh = blockIdx.x, b = bh / PNP, h = bh % PNP;
  const int wave = threadIdx.x >> 5;
  const int lane = threadIdx.x & 31, hl = lane >> 4, m = lane & 15;
  const int rowbase = blockIdx.y * 16;                  // landmark-row tile (0..3)
  const int colbase = blockIdx.z * 256 + wave * 64;
  const bf16* A = qb + ((size_t)b * NHEAD + h) * NTOK * HD + (size_t)rowbase * 64 * HD;
  const bf16* B = kb + ((size_t)b * NHEAD + h) * NTOK * HD + (size_t)colbase * HD;
  f32x8 acc[4];
#pragma unroll
  for (int t = 0; t < 4; ++t) acc[t] = zero8();
  wmma_rowtile<2>(A, HD * 64, B, HD, acc);
#pragma unroll
  for (int t = 0; t < 4; ++t)
#pragma unroll
    for (int v = 0; v < 8; ++v)
      SC[((size_t)bh * MLM + rowbase + v + 8 * hl) * NTOK + colbase + 16 * t + m] = acc[t][v];
}

// per (bh, m): rowmax of SC, extract eB (SB == SC at landmark cols), exp in place (eC)
__global__ void k_screduce(float* __restrict__ SC, float* __restrict__ eB) {
  const int bh = blockIdx.x, mrow = blockIdx.y, tid = threadIdx.x;
  float* row = SC + ((size_t)bh * MLM + mrow) * NTOK;
  __shared__ float red[256];
  float mx = -1e30f;
  for (int i = tid; i < NTOK; i += 256) mx = fmaxf(mx, row[i]);
  red[tid] = mx; __syncthreads();
  for (int s = 128; s; s >>= 1) {
    if (tid < s) red[tid] = fmaxf(red[tid], red[tid + s]);
    __syncthreads();
  }
  const float rm = red[0];
  __syncthreads();
  if (tid < MLM) {
    float sb = row[tid * 64];                       // landmark token = tid*64
    eB[((size_t)bh * MLM + mrow) * MLM + tid] = __expf(fmaxf(sb - rm, -88.0f));
  }
  __syncthreads();                                  // eB read before exp overwrite
  for (int i = tid; i < NTOK; i += 256) row[i] = __expf(row[i] - rm);
}

// per-head pinv scaling scalars (max over b, rows/cols of row/col-sums), via fp-bits atomicMax
__global__ void k_pinv_scales(const float* __restrict__ eB,
                              unsigned* __restrict__ cm, unsigned* __restrict__ rm) {
  const int bh = blockIdx.x, h = bh % PNP, t = threadIdx.x;
  const float* Bm = eB + (size_t)bh * MLM * MLM;
  float rs = 0.f, cs = 0.f;
  for (int l = 0; l < MLM; ++l) { rs += Bm[t * MLM + l]; cs += Bm[l * MLM + t]; }
  __shared__ float sA[64], sB[64];
  sA[t] = rs; sB[t] = cs; __syncthreads();
  for (int s = 32; s; s >>= 1) {
    if (t < s) { sA[t] = fmaxf(sA[t], sA[t + s]); sB[t] = fmaxf(sB[t], sB[t + s]); }
    __syncthreads();
  }
  if (t == 0) {
    atomicMax(cm + h, __float_as_uint(sA[0]));      // all values > 0
    atomicMax(rm + h, __float_as_uint(sB[0]));
  }
}

__device__ __forceinline__ void mm64(float* __restrict__ D, const float* __restrict__ A,
                                     const float* __restrict__ B, float alpha, float beta,
                                     int tid) {
  for (int p = tid; p < 4096; p += 256) {
    int i = p >> 6, j = p & 63;
    float s = 0.f;
    for (int k = 0; k < 64; ++k) s += A[i * 64 + k] * B[k * 64 + j];
    D[p] = alpha * s + ((i == j) ? beta : 0.0f);
  }
  __syncthreads();
}

// Moore-Penrose Newton iterations, 64x64 in LDS (one workgroup per (b,h))
__global__ void k_pinv(const float* __restrict__ eB, const unsigned* __restrict__ cm,
                       const unsigned* __restrict__ rm, float* __restrict__ PI) {
  __shared__ float B0[4096], B1[4096], B2[4096], B3[4096];
  const int bh = blockIdx.x, h = bh % PNP, tid = threadIdx.x;
  const float* EB = eB + (size_t)bh * 4096;
  const float inv = 1.0f / (__uint_as_float(cm[h]) * __uint_as_float(rm[h]));
  float* Z = B0; float* T1 = B1; float* T2 = B2; float* T3 = B3;
  for (int p = tid; p < 4096; p += 256) {
    int i = p >> 6, j = p & 63;
    Z[p] = EB[j * 64 + i] * inv;                    // z0 = eB^T / (cmax*rmax)
  }
  __syncthreads();
  for (int it = 0; it < 6; ++it) {
    mm64(T1, EB, Z, 1.0f, 0.0f, tid);               // xz = eB @ z
    for (int p = tid; p < 4096; p += 256) {
      int i = p >> 6, j = p & 63;
      T2[p] = ((i == j) ? 7.0f : 0.0f) - T1[p];     // 7I - xz
    }
    __syncthreads();
    mm64(T3, T1, T2, -1.0f, 15.0f, tid);            // 15I - xz@(7I-xz)
    mm64(T2, T1, T3, -1.0f, 13.0f, tid);            // 13I - xz@(...)
    mm64(T3, Z, T2, 0.25f, 0.0f, tid);              // z = 0.25 * z @ (...)
    float* tmp = Z; Z = T3; T3 = tmp;
  }
  for (int p = tid; p < 4096; p += 256) PI[(size_t)bh * 4096 + p] = Z[p];
}

// T = eC @ v_aug  (64 x 65 per bh)
__global__ void k_tmat(const float* __restrict__ eC, const bf16* __restrict__ vb,
                       float* __restrict__ Tm) {
  const int bh = blockIdx.x, b = bh / PNP, h = bh % PNP, tid = threadIdx.x;
  const float* C = eC + (size_t)bh * MLM * NTOK;
  const bf16* V = vb + ((size_t)b * NHEAD + h) * NTOK * HD;
  for (int p = tid; p < MLM * 65; p += 256) {
    int m = p / 65, d = p - m * 65;
    const float* cr = C + (size_t)m * NTOK;
    float s = 0.f;
    if (d < HD) { for (int n = 0; n < NTOK; ++n) s += cr[n] * (float)V[(size_t)n * HD + d]; }
    else        { for (int n = 0; n < NTOK; ++n) s += cr[n]; }   // ones column
    Tm[(size_t)bh * MLM * 65 + p] = s;
  }
}

// z2 = pi @ T  (64 x 65 per bh)
__global__ void k_z2(const float* __restrict__ PI, const float* __restrict__ Tm,
                     float* __restrict__ z2) {
  const int bh = blockIdx.x, tid = threadIdx.x;
  const float* P = PI + (size_t)bh * 4096;
  const float* T = Tm + (size_t)bh * MLM * 65;
  for (int p = tid; p < MLM * 65; p += 256) {
    int m = p / 65, d = p - m * 65;
    float s = 0.f;
    for (int k = 0; k < MLM; ++k) s += P[m * 64 + k] * T[k * 65 + d];
    z2[(size_t)bh * MLM * 65 + p] = s;
  }
}

// out_p rows: eA = exp(SA - rowmax); prod = eA @ z2; normalize; -> ctx bf16
__global__ void k_outp(const float* __restrict__ SA, const float* __restrict__ z2g,
                       bf16* __restrict__ ctx) {
  __shared__ float z2s[MLM * 65];
  __shared__ float ea[4][64];
  const int bh = blockIdx.x, b = bh / PNP, h = bh % PNP, tid = threadIdx.x;
  for (int p = tid; p < MLM * 65; p += 256) z2s[p] = z2g[(size_t)bh * MLM * 65 + p];
  __syncthreads();
  const int r = tid >> 6, t = tid & 63;
  const int n = blockIdx.y * 4 + r;
  const float* row = SA + ((size_t)bh * NTOK + n) * MLM;
  const float v = row[t];
  ea[r][t] = v; __syncthreads();
  for (int s = 32; s; s >>= 1) { if (t < s) ea[r][t] = fmaxf(ea[r][t], ea[r][t + s]); __syncthreads(); }
  const float mx = ea[r][0]; __syncthreads();
  const float e = __expf(v - mx);
  ea[r][t] = e; __syncthreads();
  float acc = 0.f, den = 0.f;
  for (int k = 0; k < MLM; ++k) {
    float ek = ea[r][k];
    acc += ek * z2s[k * 65 + t];
    den += ek * z2s[k * 65 + 64];
  }
  ctx[((size_t)b * NTOK + n) * DIMC + h * HD + t] = (bf16)(acc / fmaxf(den, 1e-8f));
}

// flash attention for head 11: WMMA scores + WMMA P@V with per-wave LDS relayout
__global__ void k_flash(const bf16* __restrict__ qb, const bf16* __restrict__ kb,
                        const bf16* __restrict__ vt, bf16* __restrict__ ctx) {
  __shared__ __align__(16) bf16 plds[4][16 * 32];
  const int wave = threadIdx.x >> 5;
  const int lane = threadIdx.x & 31, hl = lane >> 4, m = lane & 15;
  const int b = blockIdx.x;
  const int rowbase = blockIdx.y * 64 + wave * 16;
  const bf16* Q  = qb + (((size_t)b * NHEAD + 11) * NTOK + rowbase) * HD;
  const bf16* K  = kb + ((size_t)b * NHEAD + 11) * NTOK * HD;
  const bf16* VT = vt + (size_t)b * HD * NTOK;

  bf16x16 qa[2];
  qa[0] = load_a_frag(Q + (size_t)m * HD + 0  + hl * 16);
  qa[1] = load_a_frag(Q + (size_t)m * HD + 32 + hl * 16);

  f32x8 O[4];
#pragma unroll
  for (int t = 0; t < 4; ++t) O[t] = zero8();
  float mrun[8], lrun[8];
#pragma unroll
  for (int v = 0; v < 8; ++v) { mrun[v] = -1e30f; lrun[v] = 0.f; }

  for (int kv = 0; kv < NTOK; kv += 32) {
    // prefetch next KV tile lines into cache while this tile computes
    if (kv + 32 < NTOK) {
      __builtin_prefetch(K + (size_t)(kv + 32 + lane) * HD, 0, 3);
      __builtin_prefetch(VT + (size_t)(16 * (lane >> 3) + (lane & 7)) * NTOK + kv + 32, 0, 3);
    }
    f32x8 s[2]; s[0] = zero8(); s[1] = zero8();
#pragma unroll
    for (int t = 0; t < 2; ++t)
#pragma unroll
      for (int kk = 0; kk < 2; ++kk) {
        bf16x16 bf = *(const bf16x16*)(K + (size_t)(kv + 16 * t + m) * HD + kk * 32 + hl * 16);
        s[t] = wmma_bf16(qa[kk], bf, s[t]);
      }
    // online softmax over 32 new columns (rows live in v + 8*hl, cols across 16 lanes)
    float scale[8];
#pragma unroll
    for (int v = 0; v < 8; ++v) {
      float x = fmaxf(s[0][v], s[1][v]);
#pragma unroll
      for (int off = 8; off; off >>= 1) x = fmaxf(x, __shfl_xor(x, off, 16));
      float mn = fmaxf(mrun[v], x);
      scale[v] = __expf(mrun[v] - mn);
      mrun[v] = mn;
    }
#pragma unroll
    for (int v = 0; v < 8; ++v) {
      float p0 = __expf(s[0][v] - mrun[v]);
      float p1 = __expf(s[1][v] - mrun[v]);
      s[0][v] = p0; s[1][v] = p1;
      float rs = p0 + p1;
#pragma unroll
      for (int off = 8; off; off >>= 1) rs += __shfl_xor(rs, off, 16);
      lrun[v] = lrun[v] * scale[v] + rs;
    }
#pragma unroll
    for (int t = 0; t < 4; ++t)
#pragma unroll
      for (int v = 0; v < 8; ++v) O[t][v] *= scale[v];
    // C-layout P -> LDS -> A-layout fragment (16x32 bf16)
    bf16* pl = plds[wave];
#pragma unroll
    for (int t = 0; t < 2; ++t)
#pragma unroll
      for (int v = 0; v < 8; ++v)
        pl[(v + 8 * hl) * 32 + 16 * t + m] = (bf16)s[t][v];
    __syncthreads();
    bf16x16 pa = load_a_frag(pl + m * 32 + hl * 16);
#pragma unroll
    for (int t = 0; t < 4; ++t) {
      bf16x16 vf = *(const bf16x16*)(VT + (size_t)(16 * t + m) * NTOK + kv + hl * 16);
      O[t] = wmma_bf16(pa, vf, O[t]);
    }
    __syncthreads();
  }
#pragma unroll
  for (int t = 0; t < 4; ++t)
#pragma unroll
    for (int v = 0; v < 8; ++v) {
      int row = rowbase + v + 8 * hl;
      int d = 16 * t + m;
      ctx[((size_t)b * NTOK + row) * DIMC + 11 * HD + d] = (bf16)(O[t][v] / lrun[v]);
    }
}

// out = ctx @ proj_w^T + proj_b
__global__ void k_proj(const bf16* __restrict__ ctx, const bf16* __restrict__ W,
                       const float* __restrict__ bias, float* __restrict__ out) {
  const int wave = threadIdx.x >> 5;
  const int lane = threadIdx.x & 31, hl = lane >> 4, m = lane & 15;
  const int rowbase = blockIdx.x * 16;
  const int colbase = blockIdx.y * 256 + wave * 64;
  __builtin_prefetch(W + (size_t)(colbase + lane) * DIMC, 0, 1);
  __builtin_prefetch(W + (size_t)(colbase + 32 + lane) * DIMC, 0, 1);
  f32x8 acc[4];
#pragma unroll
  for (int t = 0; t < 4; ++t) acc[t] = zero8();
  wmma_rowtile<DIMC / 32>(ctx + (size_t)rowbase * DIMC, DIMC,
                          W + (size_t)colbase * DIMC, DIMC, acc);
#pragma unroll
  for (int t = 0; t < 4; ++t) {
    int col = colbase + 16 * t + m;
    float bs = bias[col];
#pragma unroll
    for (int v = 0; v < 8; ++v)
      out[(size_t)(rowbase + v + 8 * hl) * DIMC + col] = acc[t][v] + bs;
  }
}

// ---------------- host launch ----------------
extern "C" void kernel_launch(void* const* d_in, const int* in_sizes, int n_in,
                              void* d_out, int out_size, void* d_ws, size_t ws_size,
                              hipStream_t stream) {
  (void)in_sizes; (void)n_in; (void)out_size; (void)ws_size;
  const float* x      = (const float*)d_in[0];
  const float* qkv_w  = (const float*)d_in[1];
  const float* qkv_b  = (const float*)d_in[2];
  const float* proj_w = (const float*)d_in[3];
  const float* proj_b = (const float*)d_in[4];
  float* out = (float*)d_out;

  char* ws = (char*)d_ws;
  size_t off = 0;
  auto alloc = [&](size_t bytes) -> void* {
    void* p = ws + off;
    off += (bytes + 255) & ~(size_t)255;
    return p;
  };
  bf16*  x_bf   = (bf16*)alloc((size_t)BN * DIMC * 2);
  bf16*  qw_bf  = (bf16*)alloc((size_t)QKVN * DIMC * 2);
  bf16*  pw_bf  = (bf16*)alloc((size_t)DIMC * DIMC * 2);
  bf16*  q_bf   = (bf16*)alloc((size_t)BATCH * NHEAD * NTOK * HD * 2);
  bf16*  k_bf   = (bf16*)alloc((size_t)BATCH * NHEAD * NTOK * HD * 2);
  bf16*  v_bf   = (bf16*)alloc((size_t)BATCH * NHEAD * NTOK * HD * 2);
  bf16*  vt_bf  = (bf16*)alloc((size_t)BATCH * HD * NTOK * 2);
  float* SA     = (float*)alloc((size_t)BHP * NTOK * MLM * 4);
  float* SC     = (float*)alloc((size_t)BHP * MLM * NTOK * 4);
  float* eB     = (float*)alloc((size_t)BHP * MLM * MLM * 4);
  float* PI     = (float*)alloc((size_t)BHP * MLM * MLM * 4);
  float* Tm     = (float*)alloc((size_t)BHP * MLM * 65 * 4);
  float* z2     = (float*)alloc((size_t)BHP * MLM * 65 * 4);
  bf16*  ctx    = (bf16*)alloc((size_t)BN * DIMC * 2);
  unsigned* cm  = (unsigned*)alloc(NHEAD * 4);
  unsigned* rm  = (unsigned*)alloc(NHEAD * 4);

  hipMemsetAsync(cm, 0, 2 * NHEAD * 4, stream);   // positive-float bitmax scalars

  {
    int n = BN * DIMC;
    k_cvt_bf16<<<(n + 255) / 256, 256, 0, stream>>>(x, x_bf, n);
    n = QKVN * DIMC;
    k_cvt_bf16<<<(n + 255) / 256, 256, 0, stream>>>(qkv_w, qw_bf, n);
    n = DIMC * DIMC;
    k_cvt_bf16<<<(n + 255) / 256, 256, 0, stream>>>(proj_w, pw_bf, n);
  }
  k_qkv_gemm<<<dim3(BN / 16, QKVN / 256), 128, 0, stream>>>(x_bf, qw_bf, qkv_b,
                                                            q_bf, k_bf, v_bf, vt_bf);
  k_sa_gemm<<<dim3(BHP, NTOK / 64), 128, 0, stream>>>(q_bf, k_bf, SA);
  k_sc_gemm<<<dim3(BHP, 4, NTOK / 256), 128, 0, stream>>>(q_bf, k_bf, SC);
  k_screduce<<<dim3(BHP, MLM), 256, 0, stream>>>(SC, eB);
  k_pinv_scales<<<BHP, 64, 0, stream>>>(eB, cm, rm);
  k_pinv<<<BHP, 256, 0, stream>>>(eB, cm, rm, PI);
  k_tmat<<<BHP, 256, 0, stream>>>(SC, v_bf, Tm);
  k_z2<<<BHP, 256, 0, stream>>>(PI, Tm, z2);
  k_outp<<<dim3(BHP, NTOK / 4), 256, 0, stream>>>(SA, z2, ctx);
  k_flash<<<dim3(BATCH, NTOK / 64), 128, 0, stream>>>(q_bf, k_bf, vt_bf, ctx);
  k_proj<<<dim3(BN / 16, DIMC / 256), 128, 0, stream>>>(ctx, pw_bf, proj_b, out);
}
